// _LegacyNodeEncoder_86406152061293
// MI455X (gfx1250) — matrix operand order repaired and checked by reading, not compiled, not run on hardware
//
#include <hip/hip_runtime.h>
#include <hip/hip_bf16.h>

#define N_NODES  20000
#define N_EDGES  320000
#define DIM      512
#define N_LAYERS 3
#define LN_EPS   1e-5f

#define M_TILES   (N_NODES / 16)        // 1250
#define BLK_MT    8                     // waves per block = M-tiles per block (128 rows)
#define GRID_X    ((M_TILES + BLK_MT - 1) / BLK_MT)   // 157
#define ROWS_PAD  (GRID_X * BLK_MT * 16)              // 20096
#define KTILES    (DIM / 32)            // 16
#define NT        4                     // N-tiles per block (64 cols), shared by all waves

typedef __attribute__((ext_vector_type(16))) __bf16 v16bf;
typedef __attribute__((ext_vector_type(8)))  float  v8f;
typedef int v4i_vs __attribute__((vector_size(16)));   // matches builtin's V4i

// ------------------------------------------------------------ async LDS helpers
__device__ __forceinline__ void async_copy16(const __bf16* g, __bf16* l) {
#if __has_builtin(__builtin_amdgcn_global_load_async_to_lds_b128)
    __builtin_amdgcn_global_load_async_to_lds_b128(
        (__attribute__((address_space(1))) v4i_vs*)(uintptr_t)g,
        (__attribute__((address_space(3))) v4i_vs*)(uintptr_t)l, 0, 0);
#else
    *(uint4*)l = *(const uint4*)g;   // fallback: through-VGPR copy
#endif
}
__device__ __forceinline__ void wait_async0() {
#if __has_builtin(__builtin_amdgcn_s_wait_asynccnt)
    __builtin_amdgcn_s_wait_asynccnt(0);
#else
    asm volatile("s_wait_asynccnt 0x0" ::: "memory");
#endif
}

// ---------------------------------------------------------------- degrees
__global__ void k_degree(const int* __restrict__ src, const int* __restrict__ dst,
                         float* __restrict__ dout, float* __restrict__ din) {
    int e = blockIdx.x * blockDim.x + threadIdx.x;
    if (e < N_EDGES) {
        atomicAdd(&dout[src[e]], 1.0f);
        atomicAdd(&din[dst[e]], 1.0f);
    }
}

__global__ void k_rsqrt_clamp(float* __restrict__ a, float* __restrict__ b) {
    int i = blockIdx.x * blockDim.x + threadIdx.x;
    if (i < N_NODES) {
        a[i] = rsqrtf(fmaxf(a[i], 1.0f));
        b[i] = rsqrtf(fmaxf(b[i], 1.0f));
    }
}

// ------------------------------------------------- pack W into B-fragment layout
// Wpk layout: [layer][kt(16)][nt(32)][lane(32)][i(16)]
// lane l of tile (kt,nt): column n = nt*16 + (l&15), rows k = kt*32 + (l>>4)*8 + (i&7) + (i>>3)*16
__global__ void k_pack_w(const float* __restrict__ W, __bf16* __restrict__ Wpk) {
    int idx  = blockIdx.x * blockDim.x + threadIdx.x;   // 3*512*512 threads
    int i    =  idx        & 15;
    int lane = (idx >> 4)  & 31;
    int nt   = (idx >> 9)  & 31;
    int kt   = (idx >> 14) & 15;
    int L    =  idx >> 18;
    int n = nt * 16 + (lane & 15);
    int k = kt * 32 + (lane >> 4) * 8 + (i & 7) + (i >> 3) * 16;
    Wpk[idx] = (__bf16)W[((size_t)L * DIM + k) * DIM + n];
}

// --------------------------------- LayerNorm + affine + norm_src scale + cvt bf16
__global__ void __launch_bounds__(256)
k_ln_scale(const float* __restrict__ h, const float* __restrict__ g,
           const float* __restrict__ beta, const float* __restrict__ norm_src,
           __bf16* __restrict__ X) {
    __shared__ float s_sum[256];
    __shared__ float s_sq[256];
    int n = blockIdx.x, t = threadIdx.x;
    float x0 = h[(size_t)n * DIM + t];
    float x1 = h[(size_t)n * DIM + t + 256];
    s_sum[t] = x0 + x1;
    s_sq[t]  = x0 * x0 + x1 * x1;
    __syncthreads();
    for (int s = 128; s > 0; s >>= 1) {
        if (t < s) { s_sum[t] += s_sum[t + s]; s_sq[t] += s_sq[t + s]; }
        __syncthreads();
    }
    float mu  = s_sum[0] * (1.0f / DIM);
    float var = s_sq[0]  * (1.0f / DIM) - mu * mu;
    float r   = rsqrtf(var + LN_EPS);
    float ns  = norm_src[n];
    X[(size_t)n * DIM + t]       = (__bf16)(((x0 - mu) * r * g[t]       + beta[t])       * ns);
    X[(size_t)n * DIM + t + 256] = (__bf16)(((x1 - mu) * r * g[t + 256] + beta[t + 256]) * ns);
}

// -------------------------------------------------------------- WMMA bf16 GEMM
// Y = X @ W.  Block = 8 waves over 128 rows, sharing one 64-col group.
// B tile (4KB, contiguous in Wpk) is async-copied to LDS, double-buffered.
__global__ void __launch_bounds__(256)
k_gemm(const __bf16* __restrict__ X, const __bf16* __restrict__ Wpk,
       float* __restrict__ Y) {
    __shared__ __align__(16) __bf16 Bs[2][NT * 32 * 16];   // 2 x 4KB

    const int wave = threadIdx.x >> 5;
    const int lane = threadIdx.x & 31;
    const int mt   = blockIdx.x * BLK_MT + wave;   // M-tile of this wave (may be pad)
    const int ng   = blockIdx.y;                   // 0..7: group of 4 N-tiles

    // A-fragment addressing (16-bit A, 16x32):
    //   lanes 0-15:  row m = lane,    K = 0..7 then 16..23
    //   lanes 16-31: row m = lane-16, K = 8..15 then 24..31
    const int m    = mt * 16 + (lane & 15);
    const int koff = (lane >> 4) * 8;
    const __bf16* xrow = X + (size_t)m * DIM;

    // Per-thread async copy slice: 256 threads x 16B = 4KB = one (kt, ng) B tile.
    // Source for kt: Wpk + ((kt*32 + ng*4)*32)*16, contiguous 2048 elems.
    const __bf16* wsrc = Wpk + ((size_t)(ng * NT) * 32) * 16 + (size_t)threadIdx.x * 8;
    __bf16* lbuf[2] = { &Bs[0][threadIdx.x * 8], &Bs[1][threadIdx.x * 8] };

    v8f acc0 = {}, acc1 = {}, acc2 = {}, acc3 = {};

    // prologue: B tile for kt=0 into buffer 0; A fragment for kt=0
    async_copy16(wsrc, lbuf[0]);
    v16bf a;
    ((uint4*)&a)[0] = *(const uint4*)(xrow + koff);
    ((uint4*)&a)[1] = *(const uint4*)(xrow + koff + 16);

#pragma unroll
    for (int kt = 0; kt < KTILES; ++kt) {
        wait_async0();        // my slice of buffer kt&1 has landed
        __syncthreads();      // everyone's slice landed; buffer (kt+1)&1 is free

        if (kt + 1 < KTILES)  // launch next B tile into the other buffer
            async_copy16(wsrc + (size_t)(kt + 1) * (32 * 32 * 16), lbuf[(kt + 1) & 1]);

        v16bf a_n = a;        // software-pipeline next A fragment
        if (kt + 1 < KTILES) {
            int kb = (kt + 1) * 32;
            ((uint4*)&a_n)[0] = *(const uint4*)(xrow + kb + koff);
            ((uint4*)&a_n)[1] = *(const uint4*)(xrow + kb + koff + 16);
        }

        const __bf16* bb = &Bs[kt & 1][lane * 16];
        v16bf b0 = *(const v16bf*)(bb);
        v16bf b1 = *(const v16bf*)(bb + 1 * 32 * 16);
        v16bf b2 = *(const v16bf*)(bb + 2 * 32 * 16);
        v16bf b3 = *(const v16bf*)(bb + 3 * 32 * 16);

        acc0 = __builtin_amdgcn_wmma_f32_16x16x32_bf16(false, a, false, b0, (short)0, acc0, false, false);
        acc1 = __builtin_amdgcn_wmma_f32_16x16x32_bf16(false, a, false, b1, (short)0, acc1, false, false);
        acc2 = __builtin_amdgcn_wmma_f32_16x16x32_bf16(false, a, false, b2, (short)0, acc2, false, false);
        acc3 = __builtin_amdgcn_wmma_f32_16x16x32_bf16(false, a, false, b3, (short)0, acc3, false, false);

        a = a_n;
    }

    // C/D layout: lane l, VGPR r -> row m = (l>>4)*8 + r, col n = l&15
    if (mt < M_TILES) {   // pad tiles (uniform per wave) write nothing
        int mrow = mt * 16 + (lane >> 4) * 8;
        int ncol = ng * 64 + (lane & 15);
        float* yb = Y + (size_t)mrow * DIM + ncol;
#pragma unroll
        for (int r = 0; r < 8; ++r) {
            yb[(size_t)r * DIM +  0] = acc0[r];
            yb[(size_t)r * DIM + 16] = acc1[r];
            yb[(size_t)r * DIM + 32] = acc2[r];
            yb[(size_t)r * DIM + 48] = acc3[r];
        }
    }
}

// ------------------------------------------------------- edge gather/scatter-add
__global__ void k_scatter(const float* __restrict__ Y, const int* __restrict__ src,
                          const int* __restrict__ dst, float* __restrict__ agg) {
    int e = blockIdx.x;
    int s = src[e], d = dst[e];
    int c = threadIdx.x * 4;
    const float4 v = *(const float4*)(Y + (size_t)s * DIM + c);
    float* ap = agg + (size_t)d * DIM + c;
    atomicAdd(ap + 0, v.x);
    atomicAdd(ap + 1, v.y);
    atomicAdd(ap + 2, v.z);
    atomicAdd(ap + 3, v.w);
}

// ------------------------------------------------------- h = agg*norm_dst + b [,relu]
__global__ void __launch_bounds__(256)
k_finalize(const float* __restrict__ agg, const float* __restrict__ norm_dst,
           const float* __restrict__ bias, float* __restrict__ out, int relu) {
    int n = blockIdx.x, t = threadIdx.x;
    float nd = norm_dst[n];
#pragma unroll
    for (int j = 0; j < 2; ++j) {
        int c = t + j * 256;
        float v = agg[(size_t)n * DIM + c] * nd + bias[c];
        if (relu) v = fmaxf(v, 0.0f);
        out[(size_t)n * DIM + c] = v;
    }
}

extern "C" void kernel_launch(void* const* d_in, const int* in_sizes, int n_in,
                              void* d_out, int out_size, void* d_ws, size_t ws_size,
                              hipStream_t stream) {
    const float* feats = (const float*)d_in[0];
    const int*   src   = (const int*)d_in[1];
    const int*   dst   = (const int*)d_in[2];
    const float* ln_g  = (const float*)d_in[3];
    const float* ln_b  = (const float*)d_in[4];
    const float* W     = (const float*)d_in[5];
    const float* b     = (const float*)d_in[6];
    float* out = (float*)d_out;

    char* ws = (char*)d_ws;
    size_t off = 0;
    auto alloc = [&](size_t bytes) -> void* {
        off = (off + 255) & ~(size_t)255;
        void* p = ws + off;
        off += bytes;
        return p;
    };
    float*  norm_src = (float*) alloc((size_t)N_NODES * sizeof(float));
    float*  norm_dst = (float*) alloc((size_t)N_NODES * sizeof(float));
    __bf16* X        = (__bf16*)alloc((size_t)ROWS_PAD * DIM * sizeof(__bf16));
    __bf16* Wpk      = (__bf16*)alloc((size_t)N_LAYERS * DIM * DIM * sizeof(__bf16));
    float*  Y        = (float*) alloc((size_t)N_NODES * DIM * sizeof(float));
    float*  agg      = (float*) alloc((size_t)N_NODES * DIM * sizeof(float));

    // degrees -> norms (recomputed every call: deterministic)
    (void)hipMemsetAsync(norm_src, 0, (size_t)N_NODES * sizeof(float), stream);
    (void)hipMemsetAsync(norm_dst, 0, (size_t)N_NODES * sizeof(float), stream);
    // zero the GEMM row-padding of X (rows 20000..20095) once per launch
    (void)hipMemsetAsync(X + (size_t)N_NODES * DIM, 0,
                         (size_t)(ROWS_PAD - N_NODES) * DIM * sizeof(__bf16), stream);
    k_degree<<<(N_EDGES + 255) / 256, 256, 0, stream>>>(src, dst, norm_src, norm_dst);
    k_rsqrt_clamp<<<(N_NODES + 255) / 256, 256, 0, stream>>>(norm_src, norm_dst);

    // pack all 3 layers' W into WMMA B-fragment layout (bf16)
    k_pack_w<<<(N_LAYERS * DIM * DIM) / 256, 256, 0, stream>>>(W, Wpk);

    const float* hin = feats;
    for (int L = 0; L < N_LAYERS; ++L) {
        k_ln_scale<<<N_NODES, 256, 0, stream>>>(hin, ln_g + L * DIM, ln_b + L * DIM,
                                                norm_src, X);
        k_gemm<<<dim3(GRID_X, DIM / (NT * 16)), 256, 0, stream>>>(
            X, Wpk + (size_t)L * DIM * DIM, Y);
        (void)hipMemsetAsync(agg, 0, (size_t)N_NODES * DIM * sizeof(float), stream);
        k_scatter<<<N_EDGES, 128, 0, stream>>>(Y, src, dst, agg);
        k_finalize<<<N_NODES, 256, 0, stream>>>(agg, norm_dst, b + L * DIM, out,
                                                (L < N_LAYERS - 1) ? 1 : 0);
        hin = out;  // ping-pong through d_out
    }
}
